// GQA_60825326846195
// MI455X (gfx1250) — matrix operand order repaired and checked by reading, not compiled, not run on hardware
//
#include <hip/hip_runtime.h>
#include <hip/hip_bf16.h>
#include <stdint.h>

#define S_LEN  2048
#define E_DIM  4096
#define F_DIM  6144
#define HKV    8
#define G_SZ   4
#define D_HEAD 128
#define HG_N   32

typedef __bf16 bf16_t;
typedef __attribute__((ext_vector_type(16))) __bf16 v16bf;
typedef __attribute__((ext_vector_type(8)))  __bf16 v8bf;
typedef __attribute__((ext_vector_type(8)))  float  v8f;

union Frag { v16bf v; v8bf h[2]; };

__device__ __forceinline__ bf16_t f2bf(float f) {
  uint32_t u = __builtin_bit_cast(uint32_t, f);
  uint32_t r = (u + 0x7FFFu + ((u >> 16) & 1u)) >> 16;
  unsigned short us = (unsigned short)r;
  return __builtin_bit_cast(bf16_t, us);
}

__device__ __forceinline__ v8f wmma_bf16(v16bf a, v16bf b, v8f c) {
  // D = A(16x32) * B(32x16) + C, fp32 accumulate
  return __builtin_amdgcn_wmma_f32_16x16x32_bf16(false, a, false, b, (short)0, c,
                                                 false, false);
}

// ---------------- elementwise cast fp32 -> bf16 ----------------
__global__ void cast_bf16_kernel(const float* __restrict__ src,
                                 bf16_t* __restrict__ dst, int n) {
  int i = blockIdx.x * blockDim.x + threadIdx.x;
  if (i < n) dst[i] = f2bf(src[i]);
}

// ------------- transpose + cast: dst[c*R + r] = src[r*C + c] -------------
__global__ void transpose_bf16_kernel(const float* __restrict__ src,
                                      bf16_t* __restrict__ dst, int R, int C) {
  int i = blockIdx.x * blockDim.x + threadIdx.x;
  if (i >= R * C) return;
  int r = i / C;
  int c = i - r * C;
  dst[(size_t)c * R + r] = f2bf(src[(size_t)r * C + c]);
}

// ---------------- tiled bf16 GEMM: C(MxN,f32) = A(MxK) * Bt(NxK)^T ----------------
// Block = 256 threads = 8 waves, block tile 128x128, wave tile 64x32 (4x2 WMMA tiles).
__global__ __launch_bounds__(256)
void gemm_bf16_kernel(const bf16_t* __restrict__ A, const bf16_t* __restrict__ Bt,
                      float* __restrict__ C, int M, int N, int K) {
  const int lane = threadIdx.x & 31;
  const int wave = threadIdx.x >> 5;
  const int wm = wave >> 2;           // 0..1
  const int wn = wave & 3;            // 0..3
  const int mc = lane & 15;
  const int hl = lane >> 4;           // lane half
  const int rowBase = blockIdx.x * 128 + wm * 64;
  const int colBase = blockIdx.y * 128 + wn * 32;

  v8f acc[4][2];
#pragma unroll
  for (int i = 0; i < 4; ++i)
#pragma unroll
    for (int j = 0; j < 2; ++j) acc[i][j] = (v8f){0,0,0,0,0,0,0,0};

  const bf16_t* aptr[4];
#pragma unroll
  for (int i = 0; i < 4; ++i)
    aptr[i] = A + (size_t)(rowBase + i * 16 + mc) * K;
  const bf16_t* bptr[2];
#pragma unroll
  for (int j = 0; j < 2; ++j)
    bptr[j] = Bt + (size_t)(colBase + j * 16 + mc) * K;

  const int aoff0 = hl * 16;          // A-fragment contiguous K half
  const int aoff1 = hl * 16 + 8;
  const int boff  = hl * 16;          // B-fragment contiguous K half

  for (int k0 = 0; k0 < K; k0 += 32) {
    Frag a[4];
#pragma unroll
    for (int i = 0; i < 4; ++i) {
      a[i].h[0] = *(const v8bf*)(aptr[i] + k0 + aoff0);
      a[i].h[1] = *(const v8bf*)(aptr[i] + k0 + aoff1);
    }
    v16bf b[2];
#pragma unroll
    for (int j = 0; j < 2; ++j)
      b[j] = *(const v16bf*)(bptr[j] + k0 + boff);
#pragma unroll
    for (int i = 0; i < 4; ++i)
#pragma unroll
      for (int j = 0; j < 2; ++j)
        acc[i][j] = wmma_bf16(a[i].v, b[j], acc[i][j]);
  }

#pragma unroll
  for (int i = 0; i < 4; ++i)
#pragma unroll
    for (int j = 0; j < 2; ++j)
#pragma unroll
      for (int r = 0; r < 8; ++r)
        C[(size_t)(rowBase + i * 16 + r + 8 * hl) * N + colBase + j * 16 + mc] =
            acc[i][j][r];
}

// ---------------- RoPE split: qkv f32 -> q/k (rotated) + vT, bf16 ----------------
__global__ void rope_split_kernel(const float* __restrict__ qkv,
                                  bf16_t* __restrict__ q, bf16_t* __restrict__ k,
                                  bf16_t* __restrict__ vt) {
  int idx = blockIdx.x * blockDim.x + threadIdx.x;
  // idx -> (s, h, j, d2): d2 in [0,64), j in [0,6), h in [0,8), s in [0,2048)
  int d2 = idx & 63;
  int t = idx >> 6;
  int j = t % 6;  t /= 6;
  int h = t & 7;
  int s = t >> 3;
  if (s >= S_LEN) return;

  const float* src = qkv + (size_t)s * F_DIM + h * (6 * D_HEAD) + j * D_HEAD + 2 * d2;
  float x1 = src[0], x2 = src[1];

  if (j == 5) {  // v: cast + transpose to [h][d][t]
    bf16_t* vb = vt + (size_t)h * D_HEAD * S_LEN;
    vb[(size_t)(2 * d2) * S_LEN + s]     = f2bf(x1);
    vb[(size_t)(2 * d2 + 1) * S_LEN + s] = f2bf(x2);
    return;
  }
  float ex = (float)d2 * (2.0f / (float)D_HEAD);
  float inv_freq = __expf(-9.210340371976184f * ex);   // 10000^(-2i/D)
  float fr = (float)s * inv_freq;
  float sn, cs;
  __sincosf(fr, &sn, &cs);
  float r1 = x1 * cs - x2 * sn;
  float r2 = x1 * sn + x2 * cs;
  if (j < 4) {  // q: [hg][s][d]
    bf16_t* qb = q + (size_t)(h * G_SZ + j) * S_LEN * D_HEAD + (size_t)s * D_HEAD + 2 * d2;
    qb[0] = f2bf(r1); qb[1] = f2bf(r2);
  } else {      // k: [h][t][d]
    bf16_t* kb = k + (size_t)h * S_LEN * D_HEAD + (size_t)s * D_HEAD + 2 * d2;
    kb[0] = f2bf(r1); kb[1] = f2bf(r2);
  }
}

// ---------------- flash attention (transposed-scores trick) ----------------
// One wave per (hg, 16-row query tile). Computes S^T = K * Q^T so the softmax
// row index m lands on the lane axis: running max/sum and O rescale are
// per-lane scalars. ctx^T = V^T * P accumulated in 8 f32 WMMA tiles.
__global__ __launch_bounds__(32)
void attn_kernel(const bf16_t* __restrict__ q, const bf16_t* __restrict__ k,
                 const bf16_t* __restrict__ vt, bf16_t* __restrict__ ctx) {
  const int lane = threadIdx.x;
  const int m0 = blockIdx.x * 16;
  const int hg = blockIdx.y;
  const int h = hg >> 2;
  const int mc = lane & 15;
  const int hl = lane >> 4;
  const int aoff0 = hl * 16, aoff1 = hl * 16 + 8;  // A-fragment contiguous half
  const int boff = hl * 16;                        // B-fragment half

  // Q^T fragments (B operand): lane = m, contiguous d per lane
  const bf16_t* qbase = q + (size_t)hg * S_LEN * D_HEAD + (size_t)(m0 + mc) * D_HEAD;
  v16bf qf[4];
#pragma unroll
  for (int c = 0; c < 4; ++c)
    qf[c] = *(const v16bf*)(qbase + c * 32 + boff);

  v8f O[8];
#pragma unroll
  for (int j = 0; j < 8; ++j) O[j] = (v8f){0,0,0,0,0,0,0,0};
  float mi = -INFINITY, li = 0.f;
  const float scale = 0.08838834764831845f;  // 1/sqrt(128)
  const int mg = m0 + mc;
  const int tbEnd = (m0 + 15) >> 5;

  for (int tb = 0; tb <= tbEnd; ++tb) {
    const int t0 = tb * 32;
    v8f Sa = (v8f){0,0,0,0,0,0,0,0};
    v8f Sb = (v8f){0,0,0,0,0,0,0,0};
    const bf16_t* ka = k + (size_t)h * S_LEN * D_HEAD + (size_t)(t0 + mc) * D_HEAD;
    const bf16_t* kb = ka + (size_t)16 * D_HEAD;
#pragma unroll
    for (int c = 0; c < 4; ++c) {
      Frag fa, fb;
      fa.h[0] = *(const v8bf*)(ka + c * 32 + aoff0);
      fa.h[1] = *(const v8bf*)(ka + c * 32 + aoff1);
      fb.h[0] = *(const v8bf*)(kb + c * 32 + aoff0);
      fb.h[1] = *(const v8bf*)(kb + c * 32 + aoff1);
      Sa = wmma_bf16(fa.v, qf[c], Sa);   // rows t0..t0+15
      Sb = wmma_bf16(fb.v, qf[c], Sb);   // rows t0+16..t0+31
    }
    // scale + causal mask; C layout: vgpr r, this lane -> t = t0 + r + 8*hl (+16 for Sb)
    float p[16];
    float cmax = -1e30f;
#pragma unroll
    for (int r = 0; r < 8; ++r) {
      int ta = t0 + r + 8 * hl;
      int tbg = ta + 16;
      float sa = (ta <= mg)  ? Sa[r] * scale : -1e30f;
      float sb = (tbg <= mg) ? Sb[r] * scale : -1e30f;
      p[r] = sa; p[r + 8] = sb;
      cmax = fmaxf(cmax, fmaxf(sa, sb));
    }
    cmax = fmaxf(cmax, __shfl_xor(cmax, 16, 32));
    float mnew = fmaxf(mi, cmax);
    float corr = __expf(mi - mnew);
    mi = mnew;
    li *= corr;
#pragma unroll
    for (int j = 0; j < 8; ++j) O[j] = O[j] * corr;
    float psum = 0.f;
#pragma unroll
    for (int i = 0; i < 16; ++i) { p[i] = __expf(p[i] - mnew); psum += p[i]; }
    li += psum;
    // Repack P into B fragment (K = t): lane<16 needs t0..t0+15, lane>=16 t0+16..t0+31
    Frag pf;
#pragma unroll
    for (int r = 0; r < 8; ++r) {
      float a_o = p[r];
      float a_x = __shfl_xor(a_o, 16, 32);
      float b_o = p[r + 8];
      float b_x = __shfl_xor(b_o, 16, 32);
      pf.v[r]     = f2bf(hl ? b_x : a_o);
      pf.v[r + 8] = f2bf(hl ? b_o : a_x);
    }
    // ctx^T += V^T(d x t) * P(t x m): A operand from vT, contiguous t per lane
    const bf16_t* vb = vt + (size_t)h * D_HEAD * S_LEN + (size_t)mc * S_LEN + t0;
#pragma unroll
    for (int j = 0; j < 8; ++j) {
      Frag vf;
      const bf16_t* vr = vb + (size_t)(j * 16) * S_LEN;
      vf.h[0] = *(const v8bf*)(vr + aoff0);
      vf.h[1] = *(const v8bf*)(vr + aoff1);
      O[j] = wmma_bf16(vf.v, pf.v, O[j]);
    }
  }
  float lt = li + __shfl_xor(li, 16, 32);
  float inv = 1.0f / lt;
  // ctx^T tile (d x m): element (vgpr r, lane) -> d = j*16 + r + 8*hl, m = mc
  bf16_t* cb = ctx + (size_t)(m0 + mc) * E_DIM + hg * D_HEAD;
#pragma unroll
  for (int j = 0; j < 8; ++j)
#pragma unroll
    for (int r = 0; r < 8; ++r)
      cb[j * 16 + r + 8 * hl] = f2bf(O[j][r] * inv);
}

extern "C" void kernel_launch(void* const* d_in, const int* in_sizes, int n_in,
                              void* d_out, int out_size, void* d_ws, size_t ws_size,
                              hipStream_t stream) {
  const float* x    = (const float*)d_in[0];   // (1,2048,4096)
  const float* wqkv = (const float*)d_in[1];   // (4096,6144)
  const float* wo   = (const float*)d_in[2];   // (4096,4096)
  float* out = (float*)d_out;                  // (1,2048,4096)

  char* ws = (char*)d_ws;
  // Aliased workspace plan (~112 MB):
  //   [0,        16MB)  x_bf16            -> later ctx_bf16
  //   [16MB,     64MB+) wqkvT (50.3MB)    -> later q(16MB)+k(4.2MB)+vT(4.2MB)
  //   [67.1MB, 117.4MB) qkv f32 (50.3MB)  -> later woT (33.6MB)
  const size_t OFF_XB  = 0;
  const size_t OFF_WT  = 16777216;   // 16 MiB
  const size_t OFF_QKV = 67108864;   // 64 MiB
  bf16_t* x_bf   = (bf16_t*)(ws + OFF_XB);
  bf16_t* ctx_bf = (bf16_t*)(ws + OFF_XB);
  bf16_t* wqkvT  = (bf16_t*)(ws + OFF_WT);
  bf16_t* q_bf   = (bf16_t*)(ws + OFF_WT);
  bf16_t* k_bf   = (bf16_t*)(ws + OFF_WT + (size_t)HG_N * S_LEN * D_HEAD * 2);
  bf16_t* v_bf   = (bf16_t*)(ws + OFF_WT + (size_t)(HG_N + HKV) * S_LEN * D_HEAD * 2);
  float*  qkv    = (float*)(ws + OFF_QKV);
  bf16_t* woT    = (bf16_t*)(ws + OFF_QKV);

  // 1. cast x -> bf16
  {
    int n = S_LEN * E_DIM;
    cast_bf16_kernel<<<(n + 255) / 256, 256, 0, stream>>>(x, x_bf, n);
  }
  // 2. transpose+cast wqkv -> wqkvT (N-major for B fragments)
  {
    int n = E_DIM * F_DIM;
    transpose_bf16_kernel<<<(n + 255) / 256, 256, 0, stream>>>(wqkv, wqkvT, E_DIM, F_DIM);
  }
  // 3. qkv = x @ wqkv  (2048 x 6144, K=4096), fp32 out
  gemm_bf16_kernel<<<dim3(S_LEN / 128, F_DIM / 128), 256, 0, stream>>>(
      x_bf, wqkvT, qkv, S_LEN, F_DIM, E_DIM);
  // 4. RoPE + split into q/k/vT bf16 (overwrites wqkvT region)
  {
    int n = S_LEN * HKV * 6 * (D_HEAD / 2);
    rope_split_kernel<<<(n + 255) / 256, 256, 0, stream>>>(qkv, q_bf, k_bf, v_bf);
  }
  // 5. transpose+cast wo -> woT (overwrites qkv region)
  {
    int n = E_DIM * E_DIM;
    transpose_bf16_kernel<<<(n + 255) / 256, 256, 0, stream>>>(wo, woT, E_DIM, E_DIM);
  }
  // 6. flash attention -> ctx bf16 (overwrites x_bf region)
  attn_kernel<<<dim3(S_LEN / 16, HG_N), 32, 0, stream>>>(q_bf, k_bf, v_bf, ctx_bf);
  // 7. out = ctx @ wo
  gemm_bf16_kernel<<<dim3(S_LEN / 128, E_DIM / 128), 256, 0, stream>>>(
      ctx_bf, woT, out, S_LEN, E_DIM, E_DIM);
}
